// Horizon_attention_module_45183055954277
// MI455X (gfx1250) — compile-verified, matrix operand order checked by reading, not run on hardware
//
#include <hip/hip_runtime.h>
#include <hip/hip_bf16.h>

typedef __attribute__((ext_vector_type(16))) _Float16     v16h;
typedef __attribute__((ext_vector_type(4)))  _Float16     v4h;
typedef __attribute__((ext_vector_type(8)))  float        v8f;
typedef __attribute__((ext_vector_type(4)))  unsigned int v4u;

union ABu { v16h h; v4u q[2]; };

// ---- LDS layout (bytes) ----
// Region A (reused):  Xt[256][136]h (69632) + Wq[64][136]h (17408) + Wk (17408) + Wv[128][136]h (34816) = 139264
//                     later aliased by P[256][264]h (135168)
// Region B:           Qt[256][72]h (36864) + Kt[256][72]h (36864)  -> later aliased by OB: 8 waves * 128*18 f32 (73728)
//                     Vc[128][264]h (67584)
#define OFF_WQ   69632
#define OFF_WK   (69632 + 17408)
#define OFF_WV   (69632 + 34816)
#define OFF_QT   139264
#define OFF_KT   (139264 + 36864)
#define OFF_VC   (139264 + 73728)
#define LDS_TOTAL (139264 + 73728 + 67584)  // 280576 bytes < 320 KB

// K-dimension permutation: within each 32-element K chunk, store order
// [K0..7, K16..23, K8..15, K24..31].  With this layout one WMMA operand is
// two contiguous 16B reads per lane (ds_load_b128) instead of four
// ds_load_2addr_b32: lanes 0..15 read bytes [0,32), lanes 16..31 read [32,64)
// of each 64B row chunk, matching the ISA 7.12.2 A-operand VGPR order.
__device__ __forceinline__ int kperm(int k) {
  return (k & ~31) | (k & 7) | (((k >> 3) & 1) << 4) | (((k >> 4) & 1) << 3);
}

// Load a 16x32 f16 tile (permuted-K layout) into the WMMA A/B operand layout.
// stride is in halves and must be a multiple of 8 (16B).
__device__ __forceinline__ v16h lds_tile16(const _Float16* base, int stride) {
  const int lane = threadIdx.x & 31;
  const int m    = lane & 15;
  const int ho   = (lane >> 4) << 4;  // +16 halves (32B) for lanes 16..31
  const v4u* p = (const v4u*)(base + m * stride + ho);
  ABu t;
  t.q[0] = p[0];
  t.q[1] = p[1];
  return t.h;
}

__device__ __forceinline__ v8f wmma_f16(v16h a, v16h b, v8f c) {
  return __builtin_amdgcn_wmma_f32_16x16x32_f16(false, a, false, b, (short)0, c,
                                                false, false);
}

__global__ __launch_bounds__(256)
void Horizon_attention_module_45183055954277_kernel(
    const float* __restrict__ x,
    const float* __restrict__ wq, const float* __restrict__ gq, const float* __restrict__ bq,
    const float* __restrict__ wk, const float* __restrict__ gk, const float* __restrict__ bk,
    const float* __restrict__ wv, const float* __restrict__ bv,
    const float* __restrict__ gamma,
    float* __restrict__ out)
{
  extern __shared__ char smem[];
  _Float16* Xt = (_Float16*)smem;              // [256][136] (w-major, K=c, permuted)
  _Float16* P  = (_Float16*)smem;              // [256][264] probs (K=j, permuted), aliases Xt/W
  _Float16* Wq = (_Float16*)(smem + OFF_WQ);   // [64][136]  (K=c, permuted)
  _Float16* Wk = (_Float16*)(smem + OFF_WK);
  _Float16* Wv = (_Float16*)(smem + OFF_WV);   // [128][136]
  _Float16* Qt = (_Float16*)(smem + OFF_QT);   // [256][72]  (K=kc, permuted)
  _Float16* Kt = (_Float16*)(smem + OFF_KT);
  float*    OB = (float*)   (smem + OFF_QT);   // aliases Qt/Kt after scores
  _Float16* Vc = (_Float16*)(smem + OFF_VC);   // [128][264] (K=w, permuted)

  const int tid  = threadIdx.x;
  const int lane = tid & 31;
  const int wave = tid >> 5;
  const int nloc = lane & 15;          // N index inside a C/D tile
  const int mo   = (lane >> 4) << 3;   // M offset (0 or 8) inside a C/D tile
  const int bb   = blockIdx.x >> 7;    // batch
  const int hh   = blockIdx.x & 127;   // height row

  // ---------------- Phase 1: stage x row (f16, transposed) + weights ----------------
  for (int idx = tid; idx < 8192; idx += 256) {
    int c  = idx >> 6;
    int cp = kperm(c);
    int w4 = (idx & 63) << 2;
    const float4 f = *(const float4*)(x + ((size_t)(bb * 128 + c) * 128 + hh) * 256 + w4);
    Xt[(w4 + 0) * 136 + cp] = (_Float16)f.x;
    Xt[(w4 + 1) * 136 + cp] = (_Float16)f.y;
    Xt[(w4 + 2) * 136 + cp] = (_Float16)f.z;
    Xt[(w4 + 3) * 136 + cp] = (_Float16)f.w;
  }
  for (int idx = tid; idx < 2048; idx += 256) {
    int r  = idx >> 5;
    int c4 = (idx & 31) << 2;
    int cp = kperm(c4);                 // c4 % 8 in {0,4}: 4 consecutive stay contiguous
    float4 f = *(const float4*)(wq + r * 128 + c4);
    v4h h4;
    h4[0] = (_Float16)f.x; h4[1] = (_Float16)f.y; h4[2] = (_Float16)f.z; h4[3] = (_Float16)f.w;
    *(v4h*)(Wq + r * 136 + cp) = h4;
    f = *(const float4*)(wk + r * 128 + c4);
    h4[0] = (_Float16)f.x; h4[1] = (_Float16)f.y; h4[2] = (_Float16)f.z; h4[3] = (_Float16)f.w;
    *(v4h*)(Wk + r * 136 + cp) = h4;
  }
  for (int idx = tid; idx < 4096; idx += 256) {
    int r  = idx >> 5;
    int c4 = (idx & 31) << 2;
    int cp = kperm(c4);
    const float4 f = *(const float4*)(wv + r * 128 + c4);
    v4h h4;
    h4[0] = (_Float16)f.x; h4[1] = (_Float16)f.y; h4[2] = (_Float16)f.z; h4[3] = (_Float16)f.w;
    *(v4h*)(Wv + r * 136 + cp) = h4;
  }
  __syncthreads();

  // ---------------- Phase 2: Q/K/V projections (WMMA, fused BN / bias) ----------------
  const float rs = rsqrtf(1.0f + 1e-5f);
  for (int t = wave * 32; t < wave * 32 + 32; ++t) {
    int wt, ot, kind;
    const _Float16* W;
    if (t < 64)       { kind = 0; wt = t >> 2;         ot = t & 3;         W = Wq; }
    else if (t < 128) { kind = 1; wt = (t - 64) >> 2;  ot = (t - 64) & 3;  W = Wk; }
    else              { kind = 2; wt = (t - 128) >> 3; ot = (t - 128) & 7; W = Wv; }
    v8f acc = {};
#pragma unroll
    for (int ck = 0; ck < 4; ++ck) {
      v16h a = lds_tile16(Xt + wt * 16 * 136 + ck * 32, 136);
      v16h b = lds_tile16(W  + ot * 16 * 136 + ck * 32, 136);
      acc = wmma_f16(a, b, acc);
    }
    if (kind < 2) {
      int o  = ot * 16 + nloc;
      int op = kperm(o);                       // o is the K dim of phase 3
      float sc = (kind == 0 ? gq[o] : gk[o]) * rs;
      float bi = (kind == 0 ? bq[o] : bk[o]);
      _Float16* dst = (kind == 0 ? Qt : Kt);
#pragma unroll
      for (int r = 0; r < 8; ++r)
        dst[(wt * 16 + mo + r) * 72 + op] = (_Float16)(acc[r] * sc + bi);
    } else {
      int o = ot * 16 + nloc;
      float bi = bv[o];
#pragma unroll
      for (int r = 0; r < 8; ++r)              // w is the K dim of phase 4
        Vc[o * 264 + kperm(wt * 16 + mo + r)] = (_Float16)(acc[r] + bi);
    }
  }
  __syncthreads();

  // ---------------- Phase 3: scores (QK^T), softmax, P (f16) ----------------
  {
    const int it0 = wave * 2;
    v8f acc[2][16] = {};
#pragma unroll
    for (int kk = 0; kk < 2; ++kk) {
      v16h a0 = lds_tile16(Qt + (it0 + 0) * 16 * 72 + kk * 32, 72);
      v16h a1 = lds_tile16(Qt + (it0 + 1) * 16 * 72 + kk * 32, 72);
#pragma unroll
      for (int jt = 0; jt < 16; ++jt) {
        v16h b = lds_tile16(Kt + jt * 16 * 72 + kk * 32, 72);   // shared B
        acc[0][jt] = wmma_f16(a0, b, acc[0][jt]);
        acc[1][jt] = wmma_f16(a1, b, acc[1][jt]);
      }
    }
#pragma unroll
    for (int h = 0; h < 2; ++h) {
      const int it = it0 + h;
#pragma unroll
      for (int r = 0; r < 8; ++r) {
        float m = acc[h][0][r];
#pragma unroll
        for (int jt = 1; jt < 16; ++jt) m = fmaxf(m, acc[h][jt][r]);
#pragma unroll
        for (int s = 1; s < 16; s <<= 1) m = fmaxf(m, __shfl_xor(m, s, 16));
        float sum = 0.0f;
#pragma unroll
        for (int jt = 0; jt < 16; ++jt) {
          float e = __expf(acc[h][jt][r] - m);
          acc[h][jt][r] = e;
          sum += e;
        }
#pragma unroll
        for (int s = 1; s < 16; s <<= 1) sum += __shfl_xor(sum, s, 16);
        float inv = __builtin_amdgcn_rcpf(sum);
        int i = it * 16 + mo + r;
#pragma unroll
        for (int jt = 0; jt < 16; ++jt)        // j is the K dim of phase 4
          P[i * 264 + kperm(jt * 16 + nloc)] = (_Float16)(acc[h][jt][r] * inv);
      }
    }
  }
  __syncthreads();

  // ---------------- Phase 4: O = P * V^T, residual epilogue ----------------
  const float gam = gamma[0];
  float* ob = OB + wave * (128 * 18);  // wave-private transpose staging
  {
    const int it0 = wave * 2;
    v8f acc[2][8] = {};
#pragma unroll
    for (int jk = 0; jk < 8; ++jk) {
      v16h a0 = lds_tile16(P + (it0 + 0) * 16 * 264 + jk * 32, 264);
      v16h a1 = lds_tile16(P + (it0 + 1) * 16 * 264 + jk * 32, 264);
#pragma unroll
      for (int ct = 0; ct < 8; ++ct) {
        v16h b = lds_tile16(Vc + ct * 16 * 264 + jk * 32, 264); // shared B
        acc[0][ct] = wmma_f16(a0, b, acc[0][ct]);
        acc[1][ct] = wmma_f16(a1, b, acc[1][ct]);
      }
    }
#pragma unroll
    for (int h = 0; h < 2; ++h) {
      const int it = it0 + h;
      // stage [c][i] so global writes are contiguous along w
#pragma unroll
      for (int ct = 0; ct < 8; ++ct) {
        int c = ct * 16 + nloc;
#pragma unroll
        for (int r = 0; r < 8; ++r) ob[c * 18 + mo + r] = acc[h][ct][r];
      }
      const int i0 = it * 16;
#pragma unroll
      for (int q = 0; q < 16; ++q) {
        int idx = q * 32 + lane;    // 512 (c, float4) pairs
        int c   = idx >> 2;
        int f4  = (idx & 3) << 2;
        float o0 = ob[c * 18 + f4 + 0];
        float o1 = ob[c * 18 + f4 + 1];
        float o2 = ob[c * 18 + f4 + 2];
        float o3 = ob[c * 18 + f4 + 3];
        size_t g = ((size_t)(bb * 128 + c) * 128 + hh) * 256 + i0 + f4;
        const float4 xi = *(const float4*)(x + g);
        float4 o4;
        o4.x = gam * o0 + xi.x;
        o4.y = gam * o1 + xi.y;
        o4.z = gam * o2 + xi.z;
        o4.w = gam * o3 + xi.w;
        *(float4*)(out + g) = o4;
      }
    }
  }
}

extern "C" void kernel_launch(void* const* d_in, const int* in_sizes, int n_in,
                              void* d_out, int out_size, void* d_ws, size_t ws_size,
                              hipStream_t stream) {
  const float* x     = (const float*)d_in[0];
  const float* wq    = (const float*)d_in[1];
  const float* gq    = (const float*)d_in[2];
  const float* bq    = (const float*)d_in[3];
  const float* wk    = (const float*)d_in[4];
  const float* gk    = (const float*)d_in[5];
  const float* bk    = (const float*)d_in[6];
  const float* wv    = (const float*)d_in[7];
  const float* bv    = (const float*)d_in[8];
  const float* gamma = (const float*)d_in[9];
  float* out = (float*)d_out;

  hipFuncSetAttribute((const void*)Horizon_attention_module_45183055954277_kernel,
                      hipFuncAttributeMaxDynamicSharedMemorySize, LDS_TOTAL);

  dim3 grid(1024);   // b * h rows
  dim3 block(256);   // 8 wave32
  Horizon_attention_module_45183055954277_kernel<<<grid, block, LDS_TOTAL, stream>>>(
      x, wq, gq, bq, wk, gk, bk, wv, bv, gamma, out);
}
